// Memory_36232344109271
// MI455X (gfx1250) — matrix operand order repaired and checked by reading, not compiled, not run on hardware
//
#include <hip/hip_runtime.h>
#include <hip/hip_bf16.h>
#include <math.h>

typedef __attribute__((ext_vector_type(16))) __bf16 bf16x16;
typedef __attribute__((ext_vector_type(8)))  float   f32x8;
typedef int v4i __attribute__((vector_size(16)));

#define INV_SCALE 0.44194173824159216f  /* 1/(sqrt(512)*0.1) */
#define MMT 0.999f
#define LDSW 521

#if defined(__gfx1250__) && __has_builtin(__builtin_amdgcn_global_load_async_to_lds_b128) && __has_builtin(__builtin_amdgcn_s_wait_asynccnt)
#define USE_ASYNC_LDS 1
#else
#define USE_ASYNC_LDS 0
#endif

union AFrag { bf16x16 v; uint4 u[2]; };

#if USE_ASYNC_LDS
typedef __attribute__((address_space(1))) v4i* gv4i_p;
typedef __attribute__((address_space(3))) v4i* lv4i_p;
__device__ __forceinline__ void async_ld16(const void* g, void* l) {
  __builtin_amdgcn_global_load_async_to_lds_b128((gv4i_p)g, (lv4i_p)l, 0, 0);
}
#endif

// ---------------- prep: keys -> bf16 and bf16-transposed ----------------
__global__ void prep_keys_kernel(const float* __restrict__ keys,
                                 __bf16* __restrict__ kb, __bf16* __restrict__ kT) {
  int i = blockIdx.x * 256 + threadIdx.x;          // 512*512
  int m = i >> 9, d = i & 511;
  float v = keys[i];
  kb[i] = (__bf16)v;
  kT[d * 512 + m] = (__bf16)v;
}

// ---------------- inv_norm over D for each (b,h,w) ----------------
__global__ void norm_kernel(const float* __restrict__ query, float* __restrict__ inv_norm) {
  int n = blockIdx.x * 256 + threadIdx.x;          // 65536
  int b = n >> 12, hw = n & 4095;
  const float* base = query + (size_t)b * 2097152 + hw;
  float ss = 0.f;
  for (int d = 0; d < 512; ++d) { float v = base[(size_t)d * 4096]; ss += v * v; }
  inv_norm[n] = 1.f / fmaxf(sqrtf(ss), 1e-12f);
}

// ---------------- updated_query first half (original layout) ----------------
__global__ void uq1_kernel(const float* __restrict__ query, const float* __restrict__ inv_norm,
                           float* __restrict__ uq) {
  int i = blockIdx.x * 256 + threadIdx.x;          // 16*512*4096
  int b = i >> 21;
  int chw = i & 2097151;
  int hw = i & 4095;
  uq[(size_t)b * 4194304 + chw] = query[i] * inv_norm[b * 4096 + hw];
}

// ---------------- q (N x 512) fp32 + bf16 via LDS tile transpose ----------------
__global__ void qT_kernel(const float* __restrict__ query, const float* __restrict__ inv_norm,
                          float* __restrict__ qf, __bf16* __restrict__ qb) {
  __shared__ float tile[32][33];
  const int t = threadIdx.x;
  const int bidn = blockIdx.x >> 4, bidd = blockIdx.x & 15;
  const int n0 = bidn * 32, d0 = bidd * 32;
  const int b = n0 >> 12, hw0 = n0 & 4095;
  const int j = t & 31, i0 = t >> 5;
  const float* src = query + (size_t)b * 2097152 + (size_t)d0 * 4096 + hw0;
  for (int ii = 0; ii < 4; ++ii) {
    int i = i0 + ii * 8;
    tile[i][j] = src[(size_t)i * 4096 + j];
  }
  __syncthreads();
  const int i2 = t & 31, jr0 = t >> 5;
  for (int jj = 0; jj < 4; ++jj) {
    int jr = jr0 + jj * 8;
    int n = n0 + jr;
    float v = tile[i2][jr] * inv_norm[n];
    qf[(size_t)n * 512 + d0 + i2] = v;
    qb[(size_t)n * 512 + d0 + i2] = (__bf16)v;
  }
}

// ---------------- GEMM1: logits = q @ keys^T / scale, fused row softmax + top2 ---------
__global__ __launch_bounds__(256)
void gemm1_kernel(const __bf16* __restrict__ qb, const __bf16* __restrict__ kb,
                  float* __restrict__ logits, float* __restrict__ score_m,
                  int* __restrict__ idx0, int* __restrict__ idx1) {
  __shared__ __align__(16) char arena[81920];      // B double-buffer stage / output tile
  float* tile = (float*)arena;
  __shared__ float sv0[32][8], sv1[32][8], sp[32][8];
  __shared__ int   si0[32][8], si1[32][8];
  __shared__ float s_rmax[32], s_rsum[32];

  const int t = threadIdx.x;
  const int wave = t >> 5, lane = t & 31;
  const int rowgrp = wave >> 2, colgrp = wave & 3;
  const int n_base = blockIdx.x * 32;
  const int rowA = n_base + rowgrp * 16 + (lane & 15);
  const int koffA = (lane >> 4) * 16;
  const int koffB = (lane >> 4) * 16;

  f32x8 acc[8] = {};

#if USE_ASYNC_LDS
  for (int i = 0; i < 8; ++i) {                    // prologue: stage k-slice 0
    int chunk = i * 256 + t, col = chunk >> 2, part = chunk & 3;
    async_ld16(kb + (size_t)col * 512 + part * 8, arena + col * 80 + part * 16);
  }
#endif

  for (int ks = 0; ks < 16; ++ks) {
    const int k0 = ks * 32;
    AFrag a;
    const __bf16* ap = qb + (size_t)rowA * 512 + k0 + koffA;
    a.u[0] = *(const uint4*)ap;
    a.u[1] = *(const uint4*)(ap + 8);
#if USE_ASYNC_LDS
    if (ks < 15) {
      const int kn = (ks + 1) * 32, bi = (ks + 1) & 1;
      for (int i = 0; i < 8; ++i) {
        int chunk = i * 256 + t, col = chunk >> 2, part = chunk & 3;
        async_ld16(kb + (size_t)col * 512 + kn + part * 8,
                   arena + bi * 40960 + col * 80 + part * 16);
      }
      __builtin_amdgcn_s_wait_asynccnt(8);
    } else {
      __builtin_amdgcn_s_wait_asynccnt(0);
    }
    __syncthreads();
    const char* bbuf = arena + (ks & 1) * 40960;
#pragma unroll
    for (int jt = 0; jt < 8; ++jt) {
      const int colB = colgrp * 128 + jt * 16 + (lane & 15);
      AFrag bfr;
      const char* bp = bbuf + colB * 80 + koffB * 2;
      bfr.u[0] = *(const uint4*)bp;
      bfr.u[1] = *(const uint4*)(bp + 16);
      acc[jt] = __builtin_amdgcn_wmma_f32_16x16x32_bf16(false, a.v, false, bfr.v,
                                                        (short)0, acc[jt], false, false);
    }
    __syncthreads();
#else
#pragma unroll
    for (int jt = 0; jt < 8; ++jt) {
      const int colB = colgrp * 128 + jt * 16 + (lane & 15);
      AFrag bfr;
      const __bf16* bp = kb + (size_t)colB * 512 + k0 + koffB;
      bfr.u[0] = *(const uint4*)bp;
      bfr.u[1] = *(const uint4*)(bp + 8);
      acc[jt] = __builtin_amdgcn_wmma_f32_16x16x32_bf16(false, a.v, false, bfr.v,
                                                        (short)0, acc[jt], false, false);
    }
#endif
  }

  const int rbase = (lane >> 4) << 3;
#pragma unroll
  for (int jt = 0; jt < 8; ++jt)
#pragma unroll
    for (int i = 0; i < 8; ++i)
      tile[(rowgrp * 16 + i + rbase) * LDSW + colgrp * 128 + jt * 16 + (lane & 15)] =
          acc[jt][i] * INV_SCALE;
  __syncthreads();

  { // per-segment top2
    const int row = t >> 3, seg = t & 7, cbase = seg * 64;
    float v0 = -INFINITY, v1 = -INFINITY; int j0 = -1, j1 = -1;
    for (int c = 0; c < 64; ++c) {
      float v = tile[row * LDSW + cbase + c];
      if (v > v0) { v1 = v0; j1 = j0; v0 = v; j0 = cbase + c; }
      else if (v > v1) { v1 = v; j1 = cbase + c; }
    }
    sv0[row][seg] = v0; sv1[row][seg] = v1; si0[row][seg] = j0; si1[row][seg] = j1;
  }
  __syncthreads();
  if ((t & 7) == 0) {
    const int row = t >> 3;
    float v0 = -INFINITY, v1 = -INFINITY; int j0 = -1, j1 = -1;
    for (int s = 0; s < 8; ++s) {
      float a0 = sv0[row][s]; int b0 = si0[row][s];
      if (a0 > v0) { v1 = v0; j1 = j0; v0 = a0; j0 = b0; }
      else if (a0 > v1) { v1 = a0; j1 = b0; }
      float a1 = sv1[row][s]; int b1 = si1[row][s];
      if (a1 > v0) { v1 = v0; j1 = j0; v0 = a1; j0 = b1; }
      else if (a1 > v1) { v1 = a1; j1 = b1; }
    }
    s_rmax[row] = v0;
    idx0[n_base + row] = j0;
    idx1[n_base + row] = j1;
  }
  __syncthreads();
  {
    const int row = t >> 3, seg = t & 7, cbase = seg * 64;
    float rm = s_rmax[row], p = 0.f;
    for (int c = 0; c < 64; ++c) p += __expf(tile[row * LDSW + cbase + c] - rm);
    sp[row][seg] = p;
  }
  __syncthreads();
  if ((t & 7) == 0) {
    const int row = t >> 3; float s = 0.f;
    for (int sgi = 0; sgi < 8; ++sgi) s += sp[row][sgi];
    s_rsum[row] = s;
  }
  __syncthreads();
  for (int it = 0; it < 64; ++it) {
    int idx = it * 256 + t;
    int row = idx >> 9, col = idx & 511;
    float v = tile[row * LDSW + col];
    size_t g = (size_t)(n_base + row) * 512 + col;
    logits[g] = v;
    score_m[g] = __expf(v - s_rmax[row]) / s_rsum[row];
  }
}

// ---------------- GEMM2: concat_mem = score_m @ keys -> channel-major updated_query -----
__global__ __launch_bounds__(256)
void gemm2_kernel(const float* __restrict__ smv, const __bf16* __restrict__ kT,
                  float* __restrict__ uq) {
  __shared__ __align__(16) char arena[81920];
  float* tile = (float*)arena;
  const int t = threadIdx.x;
  const int wave = t >> 5, lane = t & 31;
  const int rowgrp = wave >> 2, colgrp = wave & 3;
  const int n_base = blockIdx.x * 32;
  const int rowA = n_base + rowgrp * 16 + (lane & 15);
  const int koffA = (lane >> 4) * 16;
  const int koffB = (lane >> 4) * 16;

  f32x8 acc[8] = {};

#if USE_ASYNC_LDS
  for (int i = 0; i < 8; ++i) {
    int chunk = i * 256 + t, col = chunk >> 2, part = chunk & 3;
    async_ld16(kT + (size_t)col * 512 + part * 8, arena + col * 80 + part * 16);
  }
#endif

  for (int ks = 0; ks < 16; ++ks) {
    const int k0 = ks * 32;
    const float* ap = smv + (size_t)rowA * 512 + k0 + koffA;
    float4 f0 = *(const float4*)ap;
    float4 f1 = *(const float4*)(ap + 4);
    float4 f2 = *(const float4*)(ap + 8);
    float4 f3 = *(const float4*)(ap + 12);
    bf16x16 av;
    av[0]=(__bf16)f0.x; av[1]=(__bf16)f0.y; av[2]=(__bf16)f0.z; av[3]=(__bf16)f0.w;
    av[4]=(__bf16)f1.x; av[5]=(__bf16)f1.y; av[6]=(__bf16)f1.z; av[7]=(__bf16)f1.w;
    av[8]=(__bf16)f2.x; av[9]=(__bf16)f2.y; av[10]=(__bf16)f2.z; av[11]=(__bf16)f2.w;
    av[12]=(__bf16)f3.x; av[13]=(__bf16)f3.y; av[14]=(__bf16)f3.z; av[15]=(__bf16)f3.w;
#if USE_ASYNC_LDS
    if (ks < 15) {
      const int kn = (ks + 1) * 32, bi = (ks + 1) & 1;
      for (int i = 0; i < 8; ++i) {
        int chunk = i * 256 + t, col = chunk >> 2, part = chunk & 3;
        async_ld16(kT + (size_t)col * 512 + kn + part * 8,
                   arena + bi * 40960 + col * 80 + part * 16);
      }
      __builtin_amdgcn_s_wait_asynccnt(8);
    } else {
      __builtin_amdgcn_s_wait_asynccnt(0);
    }
    __syncthreads();
    const char* bbuf = arena + (ks & 1) * 40960;
#pragma unroll
    for (int jt = 0; jt < 8; ++jt) {
      const int colE = colgrp * 128 + jt * 16 + (lane & 15);
      AFrag bfr;
      const char* bp = bbuf + colE * 80 + koffB * 2;
      bfr.u[0] = *(const uint4*)bp;
      bfr.u[1] = *(const uint4*)(bp + 16);
      acc[jt] = __builtin_amdgcn_wmma_f32_16x16x32_bf16(false, av, false, bfr.v,
                                                        (short)0, acc[jt], false, false);
    }
    __syncthreads();
#else
#pragma unroll
    for (int jt = 0; jt < 8; ++jt) {
      const int colE = colgrp * 128 + jt * 16 + (lane & 15);
      AFrag bfr;
      const __bf16* bp = kT + (size_t)colE * 512 + k0 + koffB;
      bfr.u[0] = *(const uint4*)bp;
      bfr.u[1] = *(const uint4*)(bp + 8);
      acc[jt] = __builtin_amdgcn_wmma_f32_16x16x32_bf16(false, av, false, bfr.v,
                                                        (short)0, acc[jt], false, false);
    }
#endif
  }

  const int rbase = (lane >> 4) << 3;
#pragma unroll
  for (int jt = 0; jt < 8; ++jt)
#pragma unroll
    for (int i = 0; i < 8; ++i)
      tile[(rowgrp * 16 + i + rbase) * LDSW + colgrp * 128 + jt * 16 + (lane & 15)] = acc[jt][i];
  __syncthreads();

  const int b = n_base >> 12;
  const int hw0 = n_base & 4095;
  float* dst = uq + (size_t)b * 4194304 + (size_t)512 * 4096 + hw0;
  for (int it = 0; it < 64; ++it) {
    int idx = it * 256 + t;
    int e = idx >> 5, jj = idx & 31;
    dst[(size_t)e * 4096 + jj] = tile[jj * LDSW + e];
  }
}

// ---------------- column softmax: per-chunk online partials, then reduce ----------------
__global__ void colpart_kernel(const float* __restrict__ logits,
                               float* __restrict__ pmax, float* __restrict__ psum) {
  const int cid = blockIdx.x, t = threadIdx.x;
  float m0 = -INFINITY, s0 = 0.f, m1 = -INFINITY, s1 = 0.f;
  for (int r = 0; r < 256; ++r) {
    const float* base = logits + (size_t)(cid * 256 + r) * 512;
    float v = base[t];
    if (v > m0) { s0 = s0 * __expf(m0 - v) + 1.f; m0 = v; } else s0 += __expf(v - m0);
    v = base[t + 256];
    if (v > m1) { s1 = s1 * __expf(m1 - v) + 1.f; m1 = v; } else s1 += __expf(v - m1);
  }
  pmax[cid * 512 + t] = m0;       psum[cid * 512 + t] = s0;
  pmax[cid * 512 + t + 256] = m1; psum[cid * 512 + t + 256] = s1;
}

__global__ void colreduce_kernel(const float* __restrict__ pmax, const float* __restrict__ psum,
                                 float* __restrict__ colmax, float* __restrict__ colsum) {
  const int col = blockIdx.x * 256 + threadIdx.x;
  float gm = -INFINITY;
  for (int i = 0; i < 256; ++i) gm = fmaxf(gm, pmax[i * 512 + col]);
  float s = 0.f;
  for (int i = 0; i < 256; ++i) s += psum[i * 512 + col] * __expf(pmax[i * 512 + col] - gm);
  colmax[col] = gm; colsum[col] = s;
}

__global__ void scoreq_kernel(float* __restrict__ buf, const float* __restrict__ colmax,
                              const float* __restrict__ colsum) {
  size_t i = (size_t)blockIdx.x * 256 + threadIdx.x;
  int col = (int)(i & 511);
  buf[i] = __expf(buf[i] - colmax[col]) / colsum[col];
}

// ---------------- losses (fixed-order two-stage reduction) ----------------
__global__ __launch_bounds__(256)
void loss_kernel(const float* __restrict__ qf, const float* __restrict__ keys,
                 const int* __restrict__ idx0, const int* __restrict__ idx1,
                 float* __restrict__ part) {
  const int t = threadIdx.x, wave = t >> 5, lane = t & 31;
  float comp = 0.f, sep = 0.f;
  for (int rr = 0; rr < 8; ++rr) {
    int n = blockIdx.x * 64 + wave * 8 + rr;
    int i0 = idx0[n], i1 = idx1[n];
    const float* qr = qf + (size_t)n * 512;
    const float* pr = keys + (size_t)i0 * 512;
    const float* nr = keys + (size_t)i1 * 512;
    float s1 = 0.f, s2 = 0.f, s3 = 0.f;
    for (int k = 0; k < 16; ++k) {
      int d = lane + k * 32;
      float qv = qr[d];
      float a = qv - pr[d];       s1 += a * a;
      float a2 = a + 1e-6f;       s2 += a2 * a2;
      float c = qv - nr[d] + 1e-6f; s3 += c * c;
    }
    for (int off = 16; off >= 1; off >>= 1) {
      s1 += __shfl_xor(s1, off); s2 += __shfl_xor(s2, off); s3 += __shfl_xor(s3, off);
    }
    comp += s1;
    sep += fmaxf(sqrtf(s2) - sqrtf(s3) + 1.f, 0.f);
  }
  __shared__ float sc[8], ssp[8];
  if (lane == 0) { sc[wave] = comp; ssp[wave] = sep; }
  __syncthreads();
  if (t == 0) {
    float C = 0.f, S = 0.f;
    for (int i = 0; i < 8; ++i) { C += sc[i]; S += ssp[i]; }
    part[blockIdx.x * 2] = C; part[blockIdx.x * 2 + 1] = S;
  }
}

__global__ void loss_final_kernel(const float* __restrict__ part, float* __restrict__ out_sc) {
  if (threadIdx.x == 0) {
    float C = 0.f, S = 0.f;
    for (int i = 0; i < 1024; ++i) { C += part[2 * i]; S += part[2 * i + 1]; }
    out_sc[0] = S / 65536.f;                 // separateness
    out_sc[1] = C / (65536.f * 512.f);       // compactness
  }
}

// ---------------- deterministic segment_sum: stable counting sort ----------------
__global__ void hist_kernel(const int* __restrict__ idx0, int* __restrict__ hist) {
  __shared__ int h[512];
  const int t = threadIdx.x, chunk = blockIdx.x;
  h[t] = 0; h[t + 256] = 0;
  __syncthreads();
  atomicAdd(&h[idx0[chunk * 256 + t]], 1);
  __syncthreads();
  hist[chunk * 512 + t] = h[t];
  hist[chunk * 512 + t + 256] = h[t + 256];
}

__global__ void offs_kernel(const int* __restrict__ hist, int* __restrict__ base2,
                            int* __restrict__ basem) {
  __shared__ int scnt[512];
  const int m = threadIdx.x;                 // 512 threads
  int run = 0;
  for (int c = 0; c < 256; ++c) {
    base2[c * 512 + m] = run;
    run += hist[c * 512 + m];
  }
  scnt[m] = run;
  __syncthreads();
  if (m == 0) {
    int acc = 0;
    for (int i = 0; i < 512; ++i) { basem[i] = acc; acc += scnt[i]; }
    basem[512] = acc;
  }
}

__global__ void place_kernel(const int* __restrict__ idx0, const int* __restrict__ base2,
                             const int* __restrict__ basem, int* __restrict__ order) {
  __shared__ int smi[256];
  const int t = threadIdx.x, chunk = blockIdx.x;
  smi[t] = idx0[chunk * 256 + t];
  __syncthreads();
  int m = smi[t], rank = 0;
  for (int r = 0; r < t; ++r) rank += (smi[r] == m);
  order[basem[m] + base2[chunk * 512 + m] + rank] = chunk * 256 + t;
}

__global__ __launch_bounds__(256)
void gather_mem_kernel(const int* __restrict__ order, const int* __restrict__ basem,
                       const float* __restrict__ score_q, const float* __restrict__ qf,
                       const float* __restrict__ keys, float* __restrict__ out_mem) {
  const int m = blockIdx.x, t = threadIdx.x;
  float acc0 = 0.f, acc1 = 0.f;
  const int s = basem[m], e = basem[m + 1];
  for (int p = s; p < e; ++p) {
    int n = order[p];
    float w = score_q[(size_t)n * 512 + m];
    acc0 += w * qf[(size_t)n * 512 + t];
    acc1 += w * qf[(size_t)n * 512 + t + 256];
  }
  float u0 = keys[(size_t)m * 512 + t] * MMT + acc0 * (1.f - MMT);
  float u1 = keys[(size_t)m * 512 + t + 256] * MMT + acc1 * (1.f - MMT);
  float ss = u0 * u0 + u1 * u1;
  for (int off = 16; off >= 1; off >>= 1) ss += __shfl_xor(ss, off);
  __shared__ float red[8];
  __shared__ float s_inv;
  if ((t & 31) == 0) red[t >> 5] = ss;
  __syncthreads();
  if (t == 0) {
    float tot = 0.f;
    for (int i = 0; i < 8; ++i) tot += red[i];
    s_inv = 1.f / fmaxf(sqrtf(tot), 1e-12f);
  }
  __syncthreads();
  out_mem[(size_t)m * 512 + t] = u0 * s_inv;
  out_mem[(size_t)m * 512 + t + 256] = u1 * s_inv;
}

extern "C" void kernel_launch(void* const* d_in, const int* in_sizes, int n_in,
                              void* d_out, int out_size, void* d_ws, size_t ws_size,
                              hipStream_t stream) {
  const float* query = (const float*)d_in[0];
  const float* keys  = (const float*)d_in[1];
  float* out = (float*)d_out;
  float* uq   = out;                     // 16,777,216
  float* memo = out + 16777216;          //    262,144
  float* sq   = out + 17039360;          // 33,554,432 (logits scratch -> score_q)
  float* smm  = out + 50593792;          // 33,554,432
  float* scal = out + 84148224;          // [sep, comp]

  char* w = (char*)d_ws;
  size_t o = 0;
  float*  inv_norm = (float*)(w + o);  o += 262144;
  float*  q_f32    = (float*)(w + o);  o += 134217728ull;
  __bf16* q_bf16   = (__bf16*)(w + o); o += 67108864ull;
  __bf16* keys_b   = (__bf16*)(w + o); o += 524288;
  __bf16* keysT_b  = (__bf16*)(w + o); o += 524288;
  int*    idx0     = (int*)(w + o);    o += 262144;
  int*    idx1     = (int*)(w + o);    o += 262144;
  float*  pmax     = (float*)(w + o);  o += 524288;
  float*  psum     = (float*)(w + o);  o += 524288;
  float*  colmax   = (float*)(w + o);  o += 2048;
  float*  colsum   = (float*)(w + o);  o += 2048;
  float*  lpart    = (float*)(w + o);  o += 8192;
  int*    hist     = (int*)(w + o);    o += 524288;
  int*    base2    = (int*)(w + o);    o += 524288;
  int*    basem    = (int*)(w + o);    o += 4096;
  int*    order    = (int*)(w + o);    o += 262144;

  prep_keys_kernel<<<1024, 256, 0, stream>>>(keys, keys_b, keysT_b);
  norm_kernel<<<256, 256, 0, stream>>>(query, inv_norm);
  uq1_kernel<<<131072, 256, 0, stream>>>(query, inv_norm, uq);
  qT_kernel<<<32768, 256, 0, stream>>>(query, inv_norm, q_f32, q_bf16);
  gemm1_kernel<<<2048, 256, 0, stream>>>(q_bf16, keys_b, sq, smm, idx0, idx1);
  gemm2_kernel<<<2048, 256, 0, stream>>>(smm, keysT_b, uq);
  colpart_kernel<<<256, 256, 0, stream>>>(sq, pmax, psum);
  colreduce_kernel<<<2, 256, 0, stream>>>(pmax, psum, colmax, colsum);
  scoreq_kernel<<<131072, 256, 0, stream>>>(sq, colmax, colsum);
  loss_kernel<<<1024, 256, 0, stream>>>(q_f32, keys, idx0, idx1, lpart);
  loss_final_kernel<<<1, 32, 0, stream>>>(lpart, scal);
  hist_kernel<<<256, 256, 0, stream>>>(idx0, hist);
  offs_kernel<<<1, 512, 0, stream>>>(hist, base2, basem);
  place_kernel<<<256, 256, 0, stream>>>(idx0, base2, basem, order);
  gather_mem_kernel<<<512, 256, 0, stream>>>(order, basem, sq, q_f32, keys, memo);
}
